// MLAAttention_4389456576827
// MI455X (gfx1250) — compile-verified, matrix operand order checked by reading, not run on hardware
//
#include <hip/hip_runtime.h>

// ---------------------------------------------------------------------------
// MLA attention for MI455X (gfx1250, wave32, WMMA bf16 16x16x32 + TDM).
// ---------------------------------------------------------------------------

typedef __bf16 bf16_t;
typedef __attribute__((ext_vector_type(16))) __bf16   v16bf;
typedef __attribute__((ext_vector_type(8)))  float    v8f;
typedef __attribute__((ext_vector_type(4)))  unsigned u32x4;
typedef __attribute__((ext_vector_type(8)))  int      i32x8;
typedef __attribute__((ext_vector_type(4)))  int      i32x4;

constexpr int DIM_   = 2048;
constexpr int H_     = 16;
constexpr int HKV_   = 4;
constexpr int ROPE_D = 64;
constexpr int NOPE_D = 128;
constexpr int V_DIM  = 128;
constexpr int QK_DIM = 192;
constexpr int KV_RANK_ = 512;
constexpr int Q_RANK_  = 1536;
constexpr int B_ = 2;
constexpr int T_ = 2048;

// A/B-fragment K index pattern for 16-bit WMMA operands (wave32):
// lanes 0-15 hold K in {0..7, 16..23}, lanes 16-31 hold K in {8..15, 24..31};
// VGPR j holds consecutive pair (k, k+1).
__device__ __forceinline__ int kpat(int j, int lh) {
  return (j < 4) ? (lh * 8 + 2 * j) : (16 + lh * 8 + 2 * (j - 4));
}

// LDS byte offset of a __shared__ object (generic AMDGCN pointer keeps the
// workgroup-relative LDS offset in its low 32 bits).
__device__ __forceinline__ unsigned lds_off(const void* p) {
  return (unsigned)(uintptr_t)p;
}

// ---------------------------------------------------------------------------
// Tensor Data Mover: async 2D tile load global -> LDS (D# per ISA 08, §8.3/8.4)
//   data_size = 2 bytes; optional LDS padding: after (2<<pad_iv) DWORDs insert
//   (pad_amt+1) DWORDs. Completion tracked by TENSORcnt.
// ---------------------------------------------------------------------------
__device__ __forceinline__ void tdm_load_2d(unsigned lds_addr, const void* gptr,
                                            unsigned tensor_d0, unsigned tensor_d1,
                                            unsigned long long stride0,
                                            unsigned tile_d0, unsigned tile_d1,
                                            int pad_iv, int pad_amt, int pad_en) {
  unsigned long long ga = (unsigned long long)(uintptr_t)gptr;
  u32x4 g0;
  g0[0] = 1u;                                            // count=1, user mode
  g0[1] = lds_addr;                                      // lds_addr [63:32]
  g0[2] = (unsigned)ga;                                  // global_addr lo
  g0[3] = (unsigned)((ga >> 32) & 0x01FFFFFFu) | (2u << 30);  // addr hi | type=2
  i32x8 g1;
  unsigned d0 = (1u << 16)                               // data_size = 2 bytes
              | (pad_en ? (1u << 20) : 0u)               // pad_enable
              | ((unsigned)pad_iv << 22)                 // pad_interval
              | ((unsigned)pad_amt << 25);               // pad_amount
  g1[0] = (int)d0;
  g1[1] = (int)((tensor_d0 & 0xFFFFu) << 16);            // tensor_dim0 lo16
  g1[2] = (int)(((tensor_d0 >> 16) & 0xFFFFu) | ((tensor_d1 & 0xFFFFu) << 16));
  g1[3] = (int)(((tensor_d1 >> 16) & 0xFFFFu) | (tile_d0 << 16));   // tile_dim0
  g1[4] = (int)(tile_d1 & 0xFFFFu);                      // tile_dim1, tile_dim2=0
  g1[5] = (int)(unsigned)(stride0 & 0xFFFFFFFFull);      // dim0_stride lo32
  g1[6] = (int)(unsigned)((stride0 >> 32) & 0xFFFFull);  // dim0_stride hi16
  g1[7] = 0;
  i32x4 z4 = {0, 0, 0, 0};
#if defined(__clang_major__) && __clang_major__ >= 23
  i32x8 z8 = {0, 0, 0, 0, 0, 0, 0, 0};
  __builtin_amdgcn_tensor_load_to_lds(g0, g1, z4, z4, z8, 0);
#else
  __builtin_amdgcn_tensor_load_to_lds(g0, g1, z4, z4, 0);
#endif
}

// ---------------------------------------------------------------------------
// fp32 -> bf16 conversion
// ---------------------------------------------------------------------------
__global__ void f32_to_bf16_kernel(const float* __restrict__ in,
                                   bf16_t* __restrict__ out, int n) {
  int i = blockIdx.x * blockDim.x + threadIdx.x;
  if (i < n) out[i] = (bf16_t)in[i];
}

// ---------------------------------------------------------------------------
// bf16 GEMM with TDM double-buffered tile staging.
// C[M,N] = A[M,K] @ B[K,N], fp32 accumulate. 256 threads = 8 waves;
// tile 128x128, K-step 32; wave = 32 rows x 64 cols (2x4 WMMA fragments).
// Wave 0 issues TDM loads for K-step t+1 while all waves compute step t.
// Requires M%128==0, N%128==0, K%32==0 (true for every GEMM here).
// ---------------------------------------------------------------------------
template <typename OutT>
__global__ __launch_bounds__(256) void gemm_bf16_kernel(
    const bf16_t* __restrict__ A, const bf16_t* __restrict__ B,
    OutT* __restrict__ C, int M, int N, int K) {
  // TDM pads rows: A 32+8 elems (80B: 20-bank stride, conflict-free for 16 rows),
  // B 128+8 elems (272B stride).
  __shared__ bf16_t As[2][128][40];
  __shared__ bf16_t Bs[2][32][136];

  const int tid  = threadIdx.x;
  const int lane = tid & 31;
  const int wave = tid >> 5;
  const int lr = lane & 15, lh = lane >> 4;
  const int m0 = blockIdx.y * 128;
  const int n0 = blockIdx.x * 128;
  const int wm = (wave >> 1) * 32;
  const int wn = (wave & 1) * 64;
  const bool issuer = (wave == 0);

  v8f acc[2][4];
#pragma unroll
  for (int mf = 0; mf < 2; ++mf)
#pragma unroll
    for (int nf = 0; nf < 4; ++nf)
#pragma unroll
      for (int e = 0; e < 8; ++e) acc[mf][nf][e] = 0.0f;

  const int nk = K >> 5;
  if (issuer) {
    // A tile t=0: [128 rows x 32], row stride K; pad 64B rows -> 80B
    tdm_load_2d(lds_off(&As[0][0][0]), A + (size_t)m0 * K,
                (unsigned)K, (unsigned)M, (unsigned long long)K, 32u, 128u,
                /*pad_iv=*/3, /*pad_amt=*/3, /*pad_en=*/1);
    // B tile t=0: [32 rows x 128], row stride N; pad 256B rows -> 272B
    tdm_load_2d(lds_off(&Bs[0][0][0]), B + n0,
                (unsigned)N, (unsigned)K, (unsigned long long)N, 128u, 32u,
                /*pad_iv=*/5, /*pad_amt=*/3, /*pad_en=*/1);
  }

  for (int kt = 0; kt < nk; ++kt) {
    if (issuer) __builtin_amdgcn_s_wait_tensorcnt(0);
    __syncthreads();  // publish buffer kt&1 to all waves

    const int buf = kt & 1;
    if (issuer && (kt + 1) < nk) {
      const int kb = (kt + 1) * 32;
      tdm_load_2d(lds_off(&As[buf ^ 1][0][0]), A + (size_t)m0 * K + kb,
                  (unsigned)K, (unsigned)M, (unsigned long long)K, 32u, 128u,
                  3, 3, 1);
      tdm_load_2d(lds_off(&Bs[buf ^ 1][0][0]), B + (size_t)kb * N + n0,
                  (unsigned)N, (unsigned)K, (unsigned long long)N, 128u, 32u,
                  5, 3, 1);
    }

    v16bf af[2], bfr[4];
#pragma unroll
    for (int mf = 0; mf < 2; ++mf) {
      const bf16_t* rowp = &As[buf][wm + mf * 16 + lr][0];
#pragma unroll
      for (int j = 0; j < 8; ++j) {
        int kk = kpat(j, lh);
        af[mf][2 * j]     = rowp[kk];
        af[mf][2 * j + 1] = rowp[kk + 1];
      }
    }
#pragma unroll
    for (int nf = 0; nf < 4; ++nf) {
      int col = wn + nf * 16 + lr;
#pragma unroll
      for (int j = 0; j < 8; ++j) {
        int kk = kpat(j, lh);
        bfr[nf][2 * j]     = Bs[buf][kk][col];
        bfr[nf][2 * j + 1] = Bs[buf][kk + 1][col];
      }
    }
#pragma unroll
    for (int mf = 0; mf < 2; ++mf)
#pragma unroll
      for (int nf = 0; nf < 4; ++nf)
        acc[mf][nf] = __builtin_amdgcn_wmma_f32_16x16x32_bf16(
            false, af[mf], false, bfr[nf], (short)0, acc[mf][nf], false, false);
    // Next iteration's top barrier separates this compute from the TDM write
    // into this buffer at step kt+2.
  }

  __syncthreads();
#pragma unroll
  for (int mf = 0; mf < 2; ++mf)
#pragma unroll
    for (int nf = 0; nf < 4; ++nf)
#pragma unroll
      for (int r = 0; r < 8; ++r) {
        int row = m0 + wm + mf * 16 + (lh ? r + 8 : r);
        int col = n0 + wn + nf * 16 + lr;
        C[(size_t)row * N + col] = (OutT)acc[mf][nf][r];
      }
}

// ---------------------------------------------------------------------------
// RoPE (in-place, bf16 buffer [B*T, nh, 64]); one thread per (row,head,pair).
// ---------------------------------------------------------------------------
__global__ void rope_kernel(bf16_t* __restrict__ buf, int nh, int total_pairs) {
  int p = blockIdx.x * blockDim.x + threadIdx.x;
  if (p >= total_pairs) return;
  int j   = p & 31;
  int rh  = p >> 5;        // row*nh + head
  int row = rh / nh;
  int t   = row % T_;
  bf16_t* x = buf + (size_t)rh * ROPE_D;
  float theta = exp2f(-(float)j * (13.287712379549449f / 32.0f));  // 10000^(-j/32)
  float ang = (float)t * theta;
  float c, s;
  __sincosf(ang, &c, &s);
  float x0 = (float)x[j];
  float x1 = (float)x[j + 32];
  x[j]      = (bf16_t)(x0 * c - x1 * s);
  x[j + 32] = (bf16_t)(x1 * c + x0 * s);
}

// ---------------------------------------------------------------------------
// Flash attention: grid (T/128, H, B), 256 threads = 8 waves.
// Each wave owns 16 q-rows; Q (192-dim) lives in registers as 6 A-fragments.
// K iterates in 64-row tiles staged in LDS; V staged transposed; P staged in
// per-wave LDS to convert C-layout -> A-layout. Online softmax in log2 domain.
// ---------------------------------------------------------------------------
__global__ __launch_bounds__(256) void mla_flash_kernel(
    const bf16_t* __restrict__ q_nope, const bf16_t* __restrict__ q_rope,
    const bf16_t* __restrict__ k_nope, const bf16_t* __restrict__ k_rope,
    const bf16_t* __restrict__ vbuf, bf16_t* __restrict__ out) {
  __shared__ bf16_t Ks[64][200];      // 64 k-rows x 192 dims (padded)
  __shared__ bf16_t Vt[V_DIM][72];    // transposed V tile (padded)
  __shared__ bf16_t Ps[128][72];      // P, 16 rows per wave (padded)

  const int tid  = threadIdx.x;
  const int lane = tid & 31;
  const int wave = tid >> 5;
  const int lr = lane & 15, lh = lane >> 4;
  const int qtile = blockIdx.x, h = blockIdx.y, b = blockIdx.z;
  const int kvh   = h >> 2;                    // GQA: 4 query heads per kv head
  const int qbase = qtile * 128;
  const size_t bT = (size_t)b * T_;

  // ---- Load Q fragments straight from global (reused across all k-tiles) ---
  v16bf qa[6];
  {
    size_t qrow = bT + qbase + wave * 16 + lr;
    const bf16_t* qn = q_nope + (qrow * H_ + h) * NOPE_D;
    const bf16_t* qr = q_rope + (qrow * H_ + h) * ROPE_D;
#pragma unroll
    for (int f = 0; f < 6; ++f)
#pragma unroll
      for (int j = 0; j < 8; ++j) {
        int d = f * 32 + kpat(j, lh);
        const bf16_t* p = (d < NOPE_D) ? (qn + d) : (qr + d - NOPE_D);
        qa[f][2 * j]     = p[0];
        qa[f][2 * j + 1] = p[1];
      }
  }

  v8f o[8];
  float m[8], l[8];
#pragma unroll
  for (int df = 0; df < 8; ++df)
#pragma unroll
    for (int e = 0; e < 8; ++e) o[df][e] = 0.0f;
#pragma unroll
  for (int r = 0; r < 8; ++r) { m[r] = -1e30f; l[r] = 0.0f; }

  const float kscale = 1.4426950408889634f / 13.856406460551018f; // log2(e)/sqrt(192)
  const int kt_end = qbase / 64 + 2;   // causal: k-tiles with kbase <= qbase+127

  for (int kt = 0; kt < kt_end; ++kt) {
    const int kbase = kt * 64;
    // ---- stage K tile (nope ++ rope) ----
    for (int idx = tid; idx < 64 * QK_DIM; idx += 256) {
      int kr_ = idx / QK_DIM, d = idx % QK_DIM;
      size_t krow = bT + kbase + kr_;
      Ks[kr_][d] = (d < NOPE_D)
          ? k_nope[(krow * HKV_ + kvh) * NOPE_D + d]
          : k_rope[(krow * HKV_ + kvh) * ROPE_D + (d - NOPE_D)];
    }
    // ---- stage V tile transposed ----
    for (int idx = tid; idx < 64 * V_DIM; idx += 256) {
      int kr_ = idx >> 7, d = idx & 127;
      Vt[d][kr_] = vbuf[((bT + kbase + kr_) * HKV_ + kvh) * V_DIM + d];
    }
    // ---- prefetch next k-tile into cache while we compute this one ----
    if (kt + 1 < kt_end) {
      size_t nrow = bT + kbase + 64 + (tid >> 2);
      const bf16_t* pn = &k_nope[(nrow * HKV_ + kvh) * NOPE_D + (tid & 3) * 32];
      __builtin_prefetch(pn, 0, 1);
      const bf16_t* pv = &vbuf[(nrow * HKV_ + kvh) * V_DIM + (tid & 3) * 32];
      __builtin_prefetch(pv, 0, 1);
      if ((tid & 3) == 0) {
        const bf16_t* pr = &k_rope[(nrow * HKV_ + kvh) * ROPE_D];
        __builtin_prefetch(pr, 0, 1);
      }
    }
    __syncthreads();

    // ---- S = Q K^T  (4 col-fragments of 16 k-positions each) ----
    float s[4][8];
#pragma unroll
    for (int nf = 0; nf < 4; ++nf) {
      v8f sf;
#pragma unroll
      for (int e = 0; e < 8; ++e) sf[e] = 0.0f;
#pragma unroll
      for (int f = 0; f < 6; ++f) {
        v16bf bfg;
        const bf16_t* rp = &Ks[nf * 16 + lr][f * 32];
#pragma unroll
        for (int j = 0; j < 8; ++j) {
          int kk = kpat(j, lh);
          bfg[2 * j]     = rp[kk];
          bfg[2 * j + 1] = rp[kk + 1];
        }
        sf = __builtin_amdgcn_wmma_f32_16x16x32_bf16(
            false, qa[f], false, bfg, (short)0, sf, false, false);
      }
      int kg = kbase + nf * 16 + lr;
#pragma unroll
      for (int r = 0; r < 8; ++r) {
        int qg = qbase + wave * 16 + (lh ? r + 8 : r);
        s[nf][r] = (kg <= qg) ? sf[r] * kscale : -1e30f;
      }
    }

    // ---- online softmax (row stats replicated across the 16-lane half) ----
    float alpha[8];
#pragma unroll
    for (int r = 0; r < 8; ++r) {
      float mx = fmaxf(fmaxf(s[0][r], s[1][r]), fmaxf(s[2][r], s[3][r]));
#pragma unroll
      for (int off = 1; off <= 8; off <<= 1)
        mx = fmaxf(mx, __shfl_xor(mx, off, 32));
      float mi = fmaxf(m[r], mx);
      alpha[r] = exp2f(m[r] - mi);
      m[r] = mi;
      float ps = 0.0f;
#pragma unroll
      for (int nf = 0; nf < 4; ++nf) {
        float p = exp2f(s[nf][r] - mi);
        s[nf][r] = p;
        ps += p;
      }
#pragma unroll
      for (int off = 1; off <= 8; off <<= 1)
        ps += __shfl_xor(ps, off, 32);
      l[r] = l[r] * alpha[r] + ps;
    }
#pragma unroll
    for (int df = 0; df < 8; ++df)
#pragma unroll
      for (int r = 0; r < 8; ++r) o[df][r] *= alpha[r];

    // ---- re-shape P through per-wave LDS (C-layout -> A-layout) ----
#pragma unroll
    for (int nf = 0; nf < 4; ++nf)
#pragma unroll
      for (int r = 0; r < 8; ++r)
        Ps[wave * 16 + (lh ? r + 8 : r)][nf * 16 + lr] = (bf16_t)s[nf][r];
    // same-wave LDS RAW: DScnt wait inserted by compiler

    // ---- O += P @ V ----
#pragma unroll
    for (int ks = 0; ks < 2; ++ks) {
      v16bf pa;
      const bf16_t* pr = &Ps[wave * 16 + lr][ks * 32];
#pragma unroll
      for (int j = 0; j < 8; ++j) {
        int kk = kpat(j, lh);
        pa[2 * j]     = pr[kk];
        pa[2 * j + 1] = pr[kk + 1];
      }
#pragma unroll
      for (int df = 0; df < 8; ++df) {
        v16bf vb;
        const bf16_t* vr = &Vt[df * 16 + lr][ks * 32];
#pragma unroll
        for (int j = 0; j < 8; ++j) {
          int kk = kpat(j, lh);
          vb[2 * j]     = vr[kk];
          vb[2 * j + 1] = vr[kk + 1];
        }
        o[df] = __builtin_amdgcn_wmma_f32_16x16x32_bf16(
            false, pa, false, vb, (short)0, o[df], false, false);
      }
    }
    __syncthreads();
  }

  // ---- normalize and write [B*T][H][V_D] bf16 ----
#pragma unroll
  for (int df = 0; df < 8; ++df)
#pragma unroll
    for (int r = 0; r < 8; ++r) {
      int row = qbase + wave * 16 + (lh ? r + 8 : r);
      int d   = df * 16 + lr;
      float val = o[df][r] / l[r];
      out[((bT + row) * H_ + h) * V_DIM + d] = (bf16_t)val;
    }
}

// ---------------------------------------------------------------------------
// Host-side orchestration
// ---------------------------------------------------------------------------
extern "C" void kernel_launch(void* const* d_in, const int* in_sizes, int n_in,
                              void* d_out, int out_size, void* d_ws, size_t ws_size,
                              hipStream_t stream) {
  (void)in_sizes; (void)n_in; (void)out_size; (void)ws_size;

  const float* x       = (const float*)d_in[0];
  const float* W_kvd   = (const float*)d_in[1];
  const float* W_kn    = (const float*)d_in[2];
  const float* W_kr    = (const float*)d_in[3];
  const float* W_v     = (const float*)d_in[4];
  const float* W_qd    = (const float*)d_in[5];
  const float* W_qn    = (const float*)d_in[6];
  const float* W_qr    = (const float*)d_in[7];
  const float* W_o     = (const float*)d_in[8];

  const int M = B_ * T_;  // 4096
  char* ws = (char*)d_ws;
  size_t off = 0;
  auto alloc_bf = [&](size_t elems) -> bf16_t* {
    bf16_t* p = (bf16_t*)(ws + off);
    off += ((elems * sizeof(bf16_t) + 255) & ~(size_t)255);
    return p;
  };

  bf16_t* xb   = alloc_bf((size_t)M * DIM_);
  bf16_t* wkvd = alloc_bf((size_t)DIM_ * KV_RANK_);
  bf16_t* wkn  = alloc_bf((size_t)KV_RANK_ * HKV_ * NOPE_D);
  bf16_t* wkr  = alloc_bf((size_t)KV_RANK_ * HKV_ * ROPE_D);
  bf16_t* wv   = alloc_bf((size_t)KV_RANK_ * HKV_ * V_DIM);
  bf16_t* wqd  = alloc_bf((size_t)DIM_ * Q_RANK_);
  bf16_t* wqn  = alloc_bf((size_t)Q_RANK_ * H_ * NOPE_D);
  bf16_t* wqr  = alloc_bf((size_t)Q_RANK_ * H_ * ROPE_D);
  bf16_t* wo   = alloc_bf((size_t)H_ * V_DIM * DIM_);
  bf16_t* kvc  = alloc_bf((size_t)M * KV_RANK_);
  bf16_t* qc   = alloc_bf((size_t)M * Q_RANK_);
  bf16_t* kn   = alloc_bf((size_t)M * HKV_ * NOPE_D);
  bf16_t* kr   = alloc_bf((size_t)M * HKV_ * ROPE_D);
  bf16_t* vv   = alloc_bf((size_t)M * HKV_ * V_DIM);
  bf16_t* qn   = alloc_bf((size_t)M * H_ * NOPE_D);
  bf16_t* qr   = alloc_bf((size_t)M * H_ * ROPE_D);
  bf16_t* ao   = alloc_bf((size_t)M * H_ * V_DIM);

  auto cvt = [&](const float* s, bf16_t* d, size_t n) {
    f32_to_bf16_kernel<<<dim3((unsigned)((n + 255) / 256)), dim3(256), 0, stream>>>(
        s, d, (int)n);
  };
  cvt(x,     xb,   (size_t)M * DIM_);
  cvt(W_kvd, wkvd, (size_t)DIM_ * KV_RANK_);
  cvt(W_kn,  wkn,  (size_t)KV_RANK_ * HKV_ * NOPE_D);
  cvt(W_kr,  wkr,  (size_t)KV_RANK_ * HKV_ * ROPE_D);
  cvt(W_v,   wv,   (size_t)KV_RANK_ * HKV_ * V_DIM);
  cvt(W_qd,  wqd,  (size_t)DIM_ * Q_RANK_);
  cvt(W_qn,  wqn,  (size_t)Q_RANK_ * H_ * NOPE_D);
  cvt(W_qr,  wqr,  (size_t)Q_RANK_ * H_ * ROPE_D);
  cvt(W_o,   wo,   (size_t)H_ * V_DIM * DIM_);

  auto gemm_bf = [&](const bf16_t* A, const bf16_t* Bm, bf16_t* Cm,
                     int Mm, int Nn, int Kk) {
    dim3 g(Nn / 128, Mm / 128);
    gemm_bf16_kernel<bf16_t><<<g, 256, 0, stream>>>(A, Bm, Cm, Mm, Nn, Kk);
  };

  // Down projections
  gemm_bf(xb, wkvd, kvc, M, KV_RANK_, DIM_);
  gemm_bf(xb, wqd,  qc,  M, Q_RANK_,  DIM_);
  // Up projections
  gemm_bf(kvc, wkn, kn, M, HKV_ * NOPE_D, KV_RANK_);
  gemm_bf(kvc, wkr, kr, M, HKV_ * ROPE_D, KV_RANK_);
  gemm_bf(kvc, wv,  vv, M, HKV_ * V_DIM,  KV_RANK_);
  gemm_bf(qc,  wqn, qn, M, H_ * NOPE_D,   Q_RANK_);
  gemm_bf(qc,  wqr, qr, M, H_ * ROPE_D,   Q_RANK_);

  // RoPE (in place)
  {
    int np = M * H_ * 32;
    rope_kernel<<<dim3((np + 255) / 256), dim3(256), 0, stream>>>(qr, H_, np);
    int nk = M * HKV_ * 32;
    rope_kernel<<<dim3((nk + 255) / 256), dim3(256), 0, stream>>>(kr, HKV_, nk);
  }

  // Flash attention
  {
    dim3 g(T_ / 128, H_, B_);
    mla_flash_kernel<<<g, 256, 0, stream>>>(qn, qr, kn, kr, vv, ao);
  }

  // Output projection -> fp32 d_out
  {
    dim3 g(DIM_ / 128, M / 128);
    gemm_bf16_kernel<float><<<g, 256, 0, stream>>>(ao, wo, (float*)d_out,
                                                   M, DIM_, H_ * V_DIM);
  }
}
